// SMPLModel_5918464934635
// MI455X (gfx1250) — matrix operand order repaired and checked by reading, not compiled, not run on hardware
//
#include <hip/hip_runtime.h>
#include <math.h>

// SMPL fused pipeline for MI455X (gfx1250, wave32, WMMA).
#define NVERT  6890
#define NJOINT 24
#define NBETA  10
#define NPOSE  207
#define NCH    (NVERT*3)        // 20670 output channels (v*3+c)
#define KDIM   224              // 207 pose + 10 shape + 7 zero pad (7 * 32)
#define KP     232              // padded LDS row stride in halves (bank-conflict free)
#define NBATCH 512

typedef _Float16 v8h  __attribute__((ext_vector_type(8)));
typedef _Float16 v16h __attribute__((ext_vector_type(16)));
typedef float    v8f  __attribute__((ext_vector_type(8)));

// Workspace byte layout
#define WS_JS_OFF    0u          // 720 f32
#define WS_JT_OFF    2880u       // 72 f32
#define WS_G_OFF     3200u       // 512*24*16 f32 = 786432 B
#define WS_A_OFF     789632u     // 512*224 f16 = 229376 B
#define WS_B16_OFF   1019008u    // 20670*224 f16 = 9260160 B
#define WS_TOTAL     10279168ull

__constant__ int d_parents[NJOINT] =
    {-1, 0, 0, 0, 1, 2, 3, 4, 5, 6, 7, 8, 9, 9, 9, 12, 13, 14, 16, 17, 18, 19, 20, 21};

// ---------------------------------------------------------------------------
// Kernel PRE: pack [posedirs | shapedirs | 0-pad] into f16 B-matrix [NCH][224].
// One block per channel row; fully coalesced.
// ---------------------------------------------------------------------------
__global__ void smpl_preconv(const float* __restrict__ posedirs,
                             const float* __restrict__ shapedirs,
                             _Float16* __restrict__ Bf16) {
  const int n = blockIdx.x;
  const int k = threadIdx.x;              // 0..223
  float v = 0.f;
  if (k < NPOSE)                 v = posedirs[(size_t)n * NPOSE + k];
  else if (k < NPOSE + NBETA)    v = shapedirs[(size_t)n * NBETA + (k - NPOSE)];
  Bf16[(size_t)n * KDIM + k] = (_Float16)v;
}

// ---------------------------------------------------------------------------
// Kernel A: JS[j,c,s] = sum_v Jr[j,v]*shapedirs[v,c,s]; JT[j,c] = sum_v Jr[j,v]*vt[v,c]
// ---------------------------------------------------------------------------
__global__ void smpl_jreg_dirs(const float* __restrict__ Jreg,
                               const float* __restrict__ shapedirs,
                               const float* __restrict__ v_template,
                               float* __restrict__ JS, float* __restrict__ JT) {
  __shared__ float sAcc[33];
  const int j = blockIdx.x;
  const int tid = threadIdx.x;
  if (tid < 33) sAcc[tid] = 0.f;
  __syncthreads();
  float acc[33];
#pragma unroll
  for (int i = 0; i < 33; ++i) acc[i] = 0.f;
  for (int v = tid; v < NVERT; v += blockDim.x) {
    float r = Jreg[j * NVERT + v];
    const float* sd = shapedirs + (size_t)v * 30;
#pragma unroll
    for (int e = 0; e < 30; ++e) acc[e] += r * sd[e];
    const float* vt = v_template + v * 3;
#pragma unroll
    for (int c = 0; c < 3; ++c) acc[30 + c] += r * vt[c];
  }
  const int lane = tid & 31;
#pragma unroll
  for (int i = 0; i < 33; ++i) {
    float val = acc[i];
#pragma unroll
    for (int o = 16; o > 0; o >>= 1) val += __shfl_down(val, o, 32);
    if (lane == 0) atomicAdd(&sAcc[i], val);   // ds_add_f32
  }
  __syncthreads();
  if (tid < 30)      JS[j * 30 + tid] = sAcc[tid];
  else if (tid < 33) JT[j * 3 + (tid - 30)] = sAcc[tid];
}

// ---------------------------------------------------------------------------
// Kernel B: per-batch rig. One wave per batch.
//   J = JT + JS*betas ; rodrigues ; kinematic chain ; G (rest-pose removed) ;
//   Abuf[b][224] = f16 [lrotmin(207) | betas(10) | 0-pad]  (WMMA A-matrix)
// ---------------------------------------------------------------------------
__global__ void smpl_rig(const float* __restrict__ betas,
                         const float* __restrict__ thetas,
                         const float* __restrict__ scale,
                         const float* __restrict__ JS,
                         const float* __restrict__ JT,
                         float* __restrict__ Gbuf,
                         _Float16* __restrict__ Abuf) {
  __shared__ float sJ[NJOINT * 3];
  __shared__ float sR[NJOINT * 9];
  __shared__ float sAw[NJOINT * 16];
  const int b = blockIdx.x;
  const int tid = threadIdx.x;
  const float* bet = betas + b * NBETA;

  if (tid < NJOINT) {
    const int j = tid;
#pragma unroll
    for (int c = 0; c < 3; ++c) {
      float a = JT[j * 3 + c];
#pragma unroll
      for (int s = 0; s < NBETA; ++s) a += JS[(j * 3 + c) * NBETA + s] * bet[s];
      sJ[j * 3 + c] = a;
    }
    const float rx = thetas[b * NJOINT * 3 + j * 3 + 0];
    const float ry = thetas[b * NJOINT * 3 + j * 3 + 1];
    const float rz = thetas[b * NJOINT * 3 + j * 3 + 2];
    const float th = sqrtf(rx * rx + ry * ry + rz * rz) + 1e-8f;
    const float inv = 1.f / th;
    const float ux = rx * inv, uy = ry * inv, uz = rz * inv;
    const float cs = cosf(th), sn = sinf(th), oc = 1.f - cs;
    float* R = sR + j * 9;
    R[0] = cs + oc * ux * ux; R[1] = oc * ux * uy - sn * uz; R[2] = oc * ux * uz + sn * uy;
    R[3] = oc * uy * ux + sn * uz; R[4] = cs + oc * uy * uy; R[5] = oc * uy * uz - sn * ux;
    R[6] = oc * uz * ux - sn * uy; R[7] = oc * uz * uy + sn * ux; R[8] = cs + oc * uz * uz;
  }
  __syncthreads();

  if (tid == 0) {
    const float sc = scale[0];
    for (int j = 0; j < NJOINT; ++j) {
      float Rl[9], t[3];
#pragma unroll
      for (int e = 0; e < 9; ++e) Rl[e] = sR[j * 9 + e];
      float* A = sAw + j * 16;
      if (j == 0) {
#pragma unroll
        for (int e = 0; e < 9; ++e) Rl[e] *= sc;
        t[0] = sJ[0]; t[1] = sJ[1]; t[2] = sJ[2];
#pragma unroll
        for (int r = 0; r < 3; ++r) {
          A[r * 4 + 0] = Rl[r * 3 + 0]; A[r * 4 + 1] = Rl[r * 3 + 1];
          A[r * 4 + 2] = Rl[r * 3 + 2]; A[r * 4 + 3] = t[r];
        }
      } else {
        const int p = d_parents[j];
#pragma unroll
        for (int c = 0; c < 3; ++c) t[c] = sJ[j * 3 + c] - sJ[p * 3 + c];
        const float* P = sAw + p * 16;
#pragma unroll
        for (int r = 0; r < 3; ++r) {
#pragma unroll
          for (int c = 0; c < 3; ++c)
            A[r * 4 + c] = P[r * 4 + 0] * Rl[0 * 3 + c] + P[r * 4 + 1] * Rl[1 * 3 + c] +
                           P[r * 4 + 2] * Rl[2 * 3 + c];
          A[r * 4 + 3] = P[r * 4 + 0] * t[0] + P[r * 4 + 1] * t[1] +
                         P[r * 4 + 2] * t[2] + P[r * 4 + 3];
        }
      }
      A[12] = 0.f; A[13] = 0.f; A[14] = 0.f; A[15] = 1.f;
    }
  }
  __syncthreads();

  if (tid < NJOINT) {
    const int j = tid;
    const float* A = sAw + j * 16;
    float* g = Gbuf + ((size_t)b * NJOINT + j) * 16;
    const float jx = sJ[j * 3], jy = sJ[j * 3 + 1], jz = sJ[j * 3 + 2];
#pragma unroll
    for (int r = 0; r < 3; ++r) {
      g[r * 4 + 0] = A[r * 4 + 0]; g[r * 4 + 1] = A[r * 4 + 1]; g[r * 4 + 2] = A[r * 4 + 2];
      g[r * 4 + 3] = A[r * 4 + 3] - (A[r * 4 + 0] * jx + A[r * 4 + 1] * jy + A[r * 4 + 2] * jz);
    }
    g[12] = 0.f; g[13] = 0.f; g[14] = 0.f; g[15] = 1.f;
  }
  // Packed WMMA A-matrix in f16 (it is consumed as f16 by the WMMA anyway).
  for (int k = tid; k < KDIM; k += blockDim.x) {
    float v = 0.f;
    if (k < NPOSE) {
      const int jj = 1 + k / 9;
      const int rc = k - (k / 9) * 9;
      v = sR[jj * 9 + rc] - ((rc == 0 || rc == 4 || rc == 8) ? 1.f : 0.f);
    } else if (k < NPOSE + NBETA) {
      v = bet[k - NPOSE];
    }
    Abuf[(size_t)b * KDIM + k] = (_Float16)v;
  }
}

// ---------------------------------------------------------------------------
// Kernel C: fused blend-shape GEMM (WMMA f16->f32) + linear blend skinning.
// Block = 192 threads (6 waves); 96 channels (=32 vertices) x 64 batches.
// v_posed = v_template + [lrotmin|betas] x [posedirs|shapedirs]^T via
// v_wmma_f32_16x16x32_f16, K = 224 = 7 steps. Then per-vertex T = sum_j w*G
// (12 live entries, exact f32), apply, add trans, write once.
// ---------------------------------------------------------------------------
__global__ __launch_bounds__(192)
void smpl_blend_skin(const _Float16* __restrict__ Abuf,
                     const _Float16* __restrict__ Bf16,
                     const float* __restrict__ posedirs,
                     const float* __restrict__ shapedirs,
                     const float* __restrict__ v_template,
                     const float* __restrict__ weights,
                     const float* __restrict__ Gbuf,
                     const float* __restrict__ trans,
                     float* __restrict__ result,
                     int preconv) {
  __shared__ __align__(16) _Float16 sB[96 * KP];   // blend-dirs tile, f16
  __shared__ __align__(16) _Float16 sA[16 * KP];   // per-batch-tile A, f16
  __shared__ float sVP[16 * 100];                  // v_posed [batch][channel]
  __shared__ float sW[32 * 24];                    // skinning weights
  __shared__ float sVt[96];                        // v_template slice (C-init)
  __shared__ float sTr[16 * 3];                    // trans slice

  const int tid = threadIdx.x;
  const int nChanBase = blockIdx.x * 96;
  const int vBase = blockIdx.x * 32;

  if (preconv) {
    // Pure f16 copy, 16B chunks: 96 rows x 28 chunks (448B/row, padded dst).
    for (int e = tid; e < 96 * 28; e += 192) {
      const int n = e / 28;
      const int c = e - n * 28;
      const int nc = nChanBase + n;
      uint4 val = make_uint4(0u, 0u, 0u, 0u);
      if (nc < NCH) val = *((const uint4*)(Bf16 + (size_t)nc * KDIM) + c);
      *((uint4*)((char*)sB + n * (KP * 2)) + c) = val;
    }
  } else {
    // Fallback: convert f32 -> f16 inline.
    for (int e = tid; e < 96 * KDIM; e += 192) {
      const int n = e / KDIM;
      const int k = e - n * KDIM;
      const int nc = nChanBase + n;
      float v = 0.f;
      if (nc < NCH) {
        if (k < NPOSE)              v = posedirs[(size_t)nc * NPOSE + k];
        else if (k < NPOSE + NBETA) v = shapedirs[(size_t)nc * NBETA + (k - NPOSE)];
      }
      sB[n * KP + k] = (_Float16)v;
    }
  }
  for (int e = tid; e < 96; e += 192) {
    const int nc = nChanBase + e;
    sVt[e] = (nc < NCH) ? v_template[nc] : 0.f;
  }
  for (int e = tid; e < 32 * 24; e += 192) {
    const int v = e / 24;
    const int vg = vBase + v;
    sW[e] = (vg < NVERT) ? weights[vg * 24 + (e - v * 24)] : 0.f;
  }
  __syncthreads();

  const int lane = tid & 31;
  const int wv = tid >> 5;          // wave id 0..5 -> channel sub-tile
  const int half16 = lane >> 4;     // 0 or 1 (half-wave)
  const int l16 = lane & 15;
  const int nTileBase = wv * 16;

  for (int bt = 0; bt < 4; ++bt) {
    const int b0 = blockIdx.y * 64 + bt * 16;
    // Stage A tile: straight f16 copy, 16 rows x 28 chunks.
    for (int e = tid; e < 16 * 28; e += 192) {
      const int m = e / 28;
      const int c = e - m * 28;
      *((uint4*)((char*)sA + m * (KP * 2)) + c) =
          *((const uint4*)(Abuf + (size_t)(b0 + m) * KDIM) + c);
    }
    for (int e = tid; e < 48; e += 192)
      sTr[e] = trans[(b0 + e / 3) * 3 + (e - (e / 3) * 3)];
    __syncthreads();

    // C init: each column n starts at v_template[n].
    const float cv = sVt[nTileBase + l16];
    v8f c = {cv, cv, cv, cv, cv, cv, cv, cv};
#pragma unroll
    for (int ks = 0; ks < 7; ++ks) {
      const int k0 = ks * 32;
      // A fragment (ISA 7.12.2, 16-bit A 16x32): lanes 0-15: K0-7,K16-23; lanes 16-31: +8
      v8h alo = *(const v8h*)&sA[l16 * KP + k0 + half16 * 8];
      v8h ahi = *(const v8h*)&sA[l16 * KP + k0 + 16 + half16 * 8];
      v16h a = __builtin_shufflevector(alo, ahi, 0, 1, 2, 3, 4, 5, 6, 7,
                                       8, 9, 10, 11, 12, 13, 14, 15);
      // B fragment: same K layout as A: lanes 0-15: K0-7,K16-23; lanes 16-31: +8
      const int brow = (nTileBase + l16) * KP + k0 + half16 * 8;
      v8h blo = *(const v8h*)&sB[brow];
      v8h bhi = *(const v8h*)&sB[brow + 16];
      v16h bf = __builtin_shufflevector(blo, bhi, 0, 1, 2, 3, 4, 5, 6, 7,
                                        8, 9, 10, 11, 12, 13, 14, 15);
      c = __builtin_amdgcn_wmma_f32_16x16x32_f16(false, a, false, bf,
                                                 (short)0, c, false, false);
    }
    // C/D layout: VGPR g -> batch row g (lanes 0-15) or g+8 (lanes 16-31).
#pragma unroll
    for (int g = 0; g < 8; ++g)
      sVP[(g + half16 * 8) * 100 + nTileBase + l16] = c[g];
    __syncthreads();

    // Per-vertex LBS: 16 batches x 32 vertices; consecutive threads ->
    // consecutive vertices for coalesced stores.
    for (int idx = tid; idx < 512; idx += 192) {
      const int vloc = idx & 31;
      const int bl = idx >> 5;
      const int vg = vBase + vloc;
      if (vg < NVERT) {
        const int bg = b0 + bl;
        const float* G = Gbuf + (size_t)bg * NJOINT * 16;
        float T[12];
#pragma unroll
        for (int e = 0; e < 12; ++e) T[e] = 0.f;
#pragma unroll
        for (int j = 0; j < NJOINT; ++j) {
          const float w = sW[vloc * 24 + j];
          const float* g = G + j * 16;
#pragma unroll
          for (int e = 0; e < 12; ++e) T[e] += w * g[e];  // bottom row is [0,0,0,1]
        }
        const float p0 = sVP[bl * 100 + vloc * 3 + 0];
        const float p1 = sVP[bl * 100 + vloc * 3 + 1];
        const float p2 = sVP[bl * 100 + vloc * 3 + 2];
        const float o0 = T[0] * p0 + T[1] * p1 + T[2] * p2 + T[3] + sTr[bl * 3 + 0];
        const float o1 = T[4] * p0 + T[5] * p1 + T[6] * p2 + T[7] + sTr[bl * 3 + 1];
        const float o2 = T[8] * p0 + T[9] * p1 + T[10] * p2 + T[11] + sTr[bl * 3 + 2];
        float* out = result + ((size_t)bg * NVERT + vg) * 3;
        out[0] = o0; out[1] = o1; out[2] = o2;
      }
    }
    __syncthreads();
  }
}

// ---------------------------------------------------------------------------
// Kernel D: joints[b,j,c] = sum_v Jr[j,v] * result[b,v,c]. Block per batch.
// ---------------------------------------------------------------------------
__global__ void smpl_joints(const float* __restrict__ Jreg,
                            const float* __restrict__ result,
                            float* __restrict__ joints) {
  __shared__ float sAcc[72];
  const int b = blockIdx.x;
  const int tid = threadIdx.x;
  if (tid < 72) sAcc[tid] = 0.f;
  __syncthreads();
  float acc[72];
#pragma unroll
  for (int i = 0; i < 72; ++i) acc[i] = 0.f;
  const float* res = result + (size_t)b * NVERT * 3;
  for (int v = tid; v < NVERT; v += blockDim.x) {
    const float p0 = res[v * 3 + 0];
    const float p1 = res[v * 3 + 1];
    const float p2 = res[v * 3 + 2];
#pragma unroll
    for (int j = 0; j < NJOINT; ++j) {
      const float r = Jreg[j * NVERT + v];
      acc[j * 3 + 0] += r * p0;
      acc[j * 3 + 1] += r * p1;
      acc[j * 3 + 2] += r * p2;
    }
  }
  const int lane = tid & 31;
#pragma unroll
  for (int i = 0; i < 72; ++i) {
    float val = acc[i];
#pragma unroll
    for (int o = 16; o > 0; o >>= 1) val += __shfl_down(val, o, 32);
    if (lane == 0) atomicAdd(&sAcc[i], val);
  }
  __syncthreads();
  if (tid < 72) joints[(size_t)b * 72 + tid] = sAcc[tid];
}

// ---------------------------------------------------------------------------
extern "C" void kernel_launch(void* const* d_in, const int* in_sizes, int n_in,
                              void* d_out, int out_size, void* d_ws, size_t ws_size,
                              hipStream_t stream) {
  const float* betas      = (const float*)d_in[0];
  const float* thetas     = (const float*)d_in[1];
  const float* trans      = (const float*)d_in[2];
  const float* scale      = (const float*)d_in[3];
  const float* v_template = (const float*)d_in[4];
  const float* shapedirs  = (const float*)d_in[5];
  const float* posedirs   = (const float*)d_in[6];
  const float* Jreg       = (const float*)d_in[7];
  const float* weights    = (const float*)d_in[8];

  float* result = (float*)d_out;                              // [512,6890,3]
  float* joints = result + (size_t)NBATCH * NVERT * 3;        // [512,24,3]

  char* ws = (char*)d_ws;
  float*     JS   = (float*)(ws + WS_JS_OFF);
  float*     JT   = (float*)(ws + WS_JT_OFF);
  float*     Gbuf = (float*)(ws + WS_G_OFF);
  _Float16*  Abuf = (_Float16*)(ws + WS_A_OFF);
  _Float16*  Bf16 = (_Float16*)(ws + WS_B16_OFF);
  const int preconv = (ws_size >= WS_TOTAL) ? 1 : 0;

  if (preconv)
    smpl_preconv<<<dim3(NCH), dim3(KDIM), 0, stream>>>(posedirs, shapedirs, Bf16);
  smpl_jreg_dirs<<<dim3(NJOINT), dim3(256), 0, stream>>>(Jreg, shapedirs, v_template, JS, JT);
  smpl_rig<<<dim3(NBATCH), dim3(32), 0, stream>>>(betas, thetas, scale, JS, JT, Gbuf, Abuf);
  const int nxBlocks = (NCH + 95) / 96;                       // 216
  smpl_blend_skin<<<dim3(nxBlocks, NBATCH / 64), dim3(192), 0, stream>>>(
      Abuf, Bf16, posedirs, shapedirs, v_template, weights, Gbuf, trans, result, preconv);
  smpl_joints<<<dim3(NBATCH), dim3(256), 0, stream>>>(Jreg, result, joints);
}
